// VoxelResBackBone8x_sub_79620103733509
// MI455X (gfx1250) — matrix operand order repaired and checked by reading, not compiled, not run on hardware
//
#include <hip/hip_runtime.h>
#include <cstdint>
#include <cstddef>

// ---------------------------------------------------------------------------
// VoxelResBackBone8x forward for MI455X (gfx1250, wave32, WMMA).
// Activations: bf16.  Accumulation + BN stats: f32.
// Conv = implicit GEMM, v_wmma_f32_16x16x32_bf16, LDS-staged A panels.
// Interior stride-1 tiles with kw*Ci % 32 == 0 use the Tensor Data Mover
// (tensor_load_to_lds + s_wait_tensorcnt) to stage the A panel.
// Workspace requirement: ~180 MB.
// ---------------------------------------------------------------------------

typedef unsigned short u16;
typedef unsigned int   u32;
typedef __attribute__((ext_vector_type(16))) __bf16 v16bf;
typedef __attribute__((ext_vector_type(8)))  float  v8f;
typedef __attribute__((ext_vector_type(4)))  int    v4i;
typedef __attribute__((ext_vector_type(8)))  int    v8i;
typedef __attribute__((ext_vector_type(4)))  unsigned int v4u;

#if defined(__has_builtin)
#if __has_builtin(__builtin_amdgcn_tensor_load_to_lds) && \
    __has_builtin(__builtin_amdgcn_s_wait_tensorcnt)
#define USE_TDM 1
#endif
#endif
#ifndef USE_TDM
#define USE_TDM 0
#endif

union Frag { v4i i[2]; v16bf v; };

__device__ __forceinline__ u16 f2bf(float f) {
  u32 u = __float_as_uint(f);
  u32 r = (u + 0x7FFFu + ((u >> 16) & 1u)) >> 16;  // round-to-nearest-even
  return (u16)r;
}
__device__ __forceinline__ float bf2f(u16 h) { return __uint_as_float(((u32)h) << 16); }

// ---------------------------------------------------------------------------
// Utility kernels
// ---------------------------------------------------------------------------
__global__ void k_zero_f32(float* p, size_t n) {
  size_t i = (size_t)blockIdx.x * blockDim.x + threadIdx.x;
  if (i < n) p[i] = 0.f;
}
__global__ void k_zero_u16(u16* p, size_t n) {
  size_t i = (size_t)blockIdx.x * blockDim.x + threadIdx.x;
  if (i < n) p[i] = 0;
}

// Scatter sparse voxels -> dense bf16 grid + f32 mask (batch == 1).
__global__ void k_scatter(const float* __restrict__ feats, const int* __restrict__ coords,
                          u16* __restrict__ dense, float* __restrict__ mask,
                          int n, int H, int W, int C) {
  int i = blockIdx.x * blockDim.x + threadIdx.x;
  if (i >= n) return;
  int z = coords[4 * i + 1], y = coords[4 * i + 2], x = coords[4 * i + 3];
  size_t vox = ((size_t)z * H + y) * W + x;
  for (int c = 0; c < C; ++c) dense[vox * C + c] = f2bf(feats[i * C + c]);
  mask[vox] = 1.f;
}

// new_mask = (conv(mask, ones, stride, pad) > 0)
__global__ void k_mask_down(const float* __restrict__ mi, float* __restrict__ mo,
                            int Di, int Hi, int Wi, int Do, int Ho, int Wo,
                            int kd, int kh, int kw, int sd, int sh, int sw,
                            int pd, int ph, int pw) {
  size_t i = (size_t)blockIdx.x * blockDim.x + threadIdx.x;
  size_t tot = (size_t)Do * Ho * Wo;
  if (i >= tot) return;
  int x = (int)(i % Wo); size_t t = i / Wo;
  int y = (int)(t % Ho); int z = (int)(t / Ho);
  float v = 0.f;
  for (int kz = 0; kz < kd; ++kz) {
    int zi = z * sd - pd + kz;
    if ((unsigned)zi >= (unsigned)Di) continue;
    for (int ky = 0; ky < kh; ++ky) {
      int yi = y * sh - ph + ky;
      if ((unsigned)yi >= (unsigned)Hi) continue;
      for (int kx = 0; kx < kw; ++kx) {
        int xi = x * sw - pw + kx;
        if ((unsigned)xi >= (unsigned)Wi) continue;
        if (mi[((size_t)zi * Hi + yi) * Wi + xi] > 0.f) v = 1.f;
      }
    }
  }
  mo[i] = v;
}

// ---------------------------------------------------------------------------
// Weight repack: (kd,kh,kw,Ci,Co) f32 -> bf16 B-fragment layout.
// Layout: wpk[((kt*NT + nt)*32 + lane)*16 + e]
//   col      = nt*16 + (lane & 15)
//   k(frag)  = (lane>>4)*16 + e   within a K=32 tile
//   K tiles per (kz,ky) plane: segTiles = ceil(kw*Ci/32), zero padded.
// ---------------------------------------------------------------------------
__global__ void k_pack_w(const float* __restrict__ w, u16* __restrict__ wpk,
                         int kd, int kh, int kw, int Ci, int Co, int segTiles) {
  int NT = Co >> 4;
  long long total = (long long)kd * kh * segTiles * NT * 512;
  long long e = (long long)blockIdx.x * blockDim.x + threadIdx.x;
  if (e >= total) return;
  int el = (int)(e & 15);
  long long t = e >> 4;
  int lane = (int)(t & 31); t >>= 5;
  int nt = (int)(t % NT);
  int kt = (int)(t / NT);
  int seg = kt / segTiles;
  int kts = kt - seg * segTiles;
  int kz = seg / kh, ky = seg - kz * kh;
  int kseg = kts * 32 + ((lane >> 4) << 4) + el;
  int col = nt * 16 + (lane & 15);
  u16 val = 0;
  if (kseg < kw * Ci) {
    int kx = kseg / Ci, c = kseg - kx * Ci;
    val = f2bf(w[(size_t)((((kz * kh + ky) * kw) + kx) * Ci + c) * Co + col]);
  }
  wpk[e] = val;
}

// ---------------------------------------------------------------------------
// Implicit-GEMM conv3d with WMMA bf16 -> f32, fused bias + output mask.
// Grid: (W-tile, Ho, Do).  Workgroup: 16 output voxels along W (M) x full Co;
// wave `nt` owns N tile nt.
// ---------------------------------------------------------------------------
struct ConvP {
  int Di, Hi, Wi, Ci, Do, Ho, Wo, Co;
  int kd, kh, kw, sd, sh, sw, pd, ph, pw;
  int segTiles, NT;
};

__global__ __launch_bounds__(256) void k_conv_wmma(
    const u16* __restrict__ in, const u16* __restrict__ wpk,
    const float* __restrict__ bias, const float* __restrict__ omask,
    float* __restrict__ out, ConvP P) {
  __shared__ __attribute__((aligned(16))) u16 sA[16 * 384];  // 12 KB panel

  const int segKp = P.segTiles << 5;
  const int segK  = P.kw * P.Ci;

  int oy = blockIdx.y, oz = blockIdx.z;
  int x0 = blockIdx.x << 4;

  int lane  = threadIdx.x & 31;
  int nt    = threadIdx.x >> 5;
  int row   = lane & 15;
  int khalf = (lane >> 4) << 4;

  __builtin_prefetch(wpk, 0, 0);  // global_prefetch_b8 on the weight panel

  v8f acc = {0.f, 0.f, 0.f, 0.f, 0.f, 0.f, 0.f, 0.f};

  for (int kz = 0; kz < P.kd; ++kz) {
    int zi = oz * P.sd - P.pd + kz;
    for (int ky = 0; ky < P.kh; ++ky) {
      int yi = oy * P.sh - P.ph + ky;
      bool planeOK = ((unsigned)zi < (unsigned)P.Di) && ((unsigned)yi < (unsigned)P.Hi);
      const u16* inP = in + ((size_t)zi * P.Hi + yi) * (size_t)P.Wi * P.Ci;

      __syncthreads();  // previous iteration's WMMA reads done

#if USE_TDM
      // TDM fast path: stride-1, dense K (kw*Ci % 32 == 0), fully interior
      // tile -> 2D tensor tile, 16 rows of segK contiguous bf16, row stride
      // Ci elements (overlapping reads expressed by tensor_dim0_stride).
      bool tdmOK = planeOK && (P.sw == 1) && (segK == segKp) &&
                   ((x0 - P.pw) >= 0) && ((x0 + 15 + P.kw - 1 - P.pw) < P.Wi);
      if (tdmOK) {
        if (threadIdx.x == 0) {
          unsigned long long gaddr =
              (unsigned long long)(const char*)(inP + (size_t)(x0 - P.pw) * P.Ci);
          u32 ldsoff = (u32)(uintptr_t)(&sA[0]);  // flat LDS addr[31:0] = offset
          v4u g0;
          g0.x = 1u;                                   // count=1, user D#
          g0.y = ldsoff;                               // lds_addr
          g0.z = (u32)gaddr;                           // global_addr[31:0]
          g0.w = (u32)((gaddr >> 32) & 0x01FFFFFFull)  // global_addr[56:32]
                 | (2u << 30);                         // type = 2 ("image")
          v8i g1;
          g1[0] = (int)(1u << 16);                     // data_size = 2 bytes
          g1[1] = (int)(((u32)segK & 0xFFFFu) << 16);  // tensor_dim0[15:0]
          g1[2] = (int)(((u32)segK >> 16) | (16u << 16)); // dim0 hi | tensor_dim1=16
          g1[3] = (int)((u32)segK << 16);              // dim1 hi(0) | tile_dim0=segK
          g1[4] = 16;                                  // tile_dim1=16, tile_dim2=0
          g1[5] = (int)(u32)P.Ci;                      // tensor_dim0_stride = Ci
          g1[6] = 0;
          g1[7] = 0;
          v4i g2 = {0, 0, 0, 0}, g3 = {0, 0, 0, 0};    // <=2D tensor
          v8i g4 = {0, 0, 0, 0, 0, 0, 0, 0};           // unused extra group
          __builtin_amdgcn_tensor_load_to_lds(g0, g1, g2, g3, g4, 0);
        }
        if (threadIdx.x < 32) __builtin_amdgcn_s_wait_tensorcnt(0);
        __syncthreads();
      } else
#endif
      {
        if ((P.Ci & 7) == 0) {          // vector fill: 16B LDS stores
          int g8 = segKp >> 3;
          for (int g = threadIdx.x; g < 16 * g8; g += blockDim.x) {
            int m = g / g8; int k8 = (g - m * g8) << 3;
            v4i v = {0, 0, 0, 0};
            if (planeOK && k8 < segK) {
              int kx = k8 / P.Ci; int c = k8 - kx * P.Ci;
              int xi = (x0 + m) * P.sw - P.pw + kx;
              if ((unsigned)xi < (unsigned)P.Wi)
                v = *(const v4i*)(inP + (size_t)xi * P.Ci + c);
            }
            *(v4i*)(sA + m * segKp + k8) = v;
          }
        } else {                        // scalar fill (stem: Ci == 5)
          for (int g = threadIdx.x; g < 16 * segKp; g += blockDim.x) {
            int m = g / segKp; int k = g - m * segKp;
            u16 v = 0;
            if (planeOK && k < segK) {
              int kx = k / P.Ci; int c = k - kx * P.Ci;
              int xi = (x0 + m) * P.sw - P.pw + kx;
              if ((unsigned)xi < (unsigned)P.Wi)
                v = inP[(size_t)xi * P.Ci + c];
            }
            sA[m * segKp + k] = v;
          }
        }
        __syncthreads();
      }

      int ktbase = (kz * P.kh + ky) * P.segTiles;
      for (int t = 0; t < P.segTiles; ++t) {
        Frag A, B;
        const u16* ap = sA + row * segKp + (t << 5) + khalf;
        A.i[0] = *(const v4i*)ap;          // K khalf .. khalf+7
        A.i[1] = *(const v4i*)(ap + 8);    // K khalf+8 .. khalf+15
        const u16* bp = wpk + ((((size_t)(ktbase + t)) * P.NT + nt) * 32 + lane) * 16;
        B.i[0] = *(const v4i*)bp;
        B.i[1] = *(const v4i*)(bp + 8);
        acc = __builtin_amdgcn_wmma_f32_16x16x32_bf16(
            false, A.v, false, B.v, (short)0, acc, false, false);
      }
    }
  }

  // D layout: VGPR r -> M = (lane>=16)*8 + r, N = lane & 15
  int n  = lane & 15;
  int co = (nt << 4) + n;
  float bv = bias ? bias[co] : 0.f;
  int mb = (lane >> 4) << 3;
  size_t rowbase = ((size_t)oz * P.Ho + oy) * (size_t)P.Wo;
  for (int r = 0; r < 8; ++r) {
    int x = x0 + mb + r;
    if (x < P.Wo) {
      size_t vox = rowbase + x;
      out[vox * P.Co + co] = (acc[r] + bv) * omask[vox];
    }
  }
}

// ---------------------------------------------------------------------------
// Masked BN: one-pass stats (sum, sumsq, count) via LDS + global f32 atomics.
// stats layout: [0..C) sum(x*m), [C..2C) sum((x*m)^2), [2C] count.
// ---------------------------------------------------------------------------
__global__ void k_bn_stats(const float* __restrict__ x, const float* __restrict__ mask,
                           float* __restrict__ stats, size_t nvox, int C) {
  __shared__ float s_sum[128], s_sq[128];
  __shared__ float s_cnt;
  for (int c = threadIdx.x; c < C; c += blockDim.x) { s_sum[c] = 0.f; s_sq[c] = 0.f; }
  if (threadIdx.x == 0) s_cnt = 0.f;
  __syncthreads();

  size_t tid = (size_t)blockIdx.x * blockDim.x + threadIdx.x;
  size_t stride = (size_t)gridDim.x * blockDim.x;
  size_t total = nvox * (size_t)C;
  for (size_t e = tid; e < total; e += stride) {
    int c = (int)(e % (size_t)C);
    float mv = mask[e / (size_t)C];
    float v = x[e] * mv;
    atomicAdd(&s_sum[c], v);
    atomicAdd(&s_sq[c], v * v);
  }
  for (size_t v = tid; v < nvox; v += stride)
    if (mask[v] != 0.f) atomicAdd(&s_cnt, 1.f);
  __syncthreads();

  for (int c = threadIdx.x; c < C; c += blockDim.x) {
    atomicAdd(&stats[c], s_sum[c]);
    atomicAdd(&stats[C + c], s_sq[c]);
  }
  if (threadIdx.x == 0) atomicAdd(&stats[2 * C], s_cnt);
}

// Apply BN (+ optional residual, ReLU); emit bf16 act or f32 final output.
__global__ void k_bn_apply(const float* __restrict__ x, const float* __restrict__ mask,
                           const float* __restrict__ stats, const float* __restrict__ g,
                           const float* __restrict__ b, const u16* __restrict__ resid,
                           u16* __restrict__ out_bf, float* __restrict__ out_f32,
                           size_t nvox, int C, int relu, float eps) {
  size_t e = (size_t)blockIdx.x * blockDim.x + threadIdx.x;
  if (e >= nvox * (size_t)C) return;
  int c = (int)(e % (size_t)C);
  size_t v = e / (size_t)C;
  float cnt = stats[2 * C]; cnt = cnt < 1.f ? 1.f : cnt;
  float mean = stats[c] / cnt;
  float var  = stats[C + c] / cnt - mean * mean;
  float mv = mask[v];
  float y = (x[e] - mean) * mv * rsqrtf(var + eps) * g[c] + b[c] * mv;
  if (resid) y += bf2f(resid[e]);
  if (relu) y = y > 0.f ? y : 0.f;
  if (out_f32) out_f32[e] = y;
  else         out_bf[e] = f2bf(y);
}

// ---------------------------------------------------------------------------
// Host orchestration
// ---------------------------------------------------------------------------
struct Dims { int D, H, W, C; };

extern "C" void kernel_launch(void* const* d_in, const int* in_sizes, int n_in,
                              void* d_out, int out_size, void* d_ws, size_t ws_size,
                              hipStream_t stream) {
  (void)in_sizes; (void)n_in; (void)out_size; (void)ws_size;

  const float* feats  = (const float*)d_in[0];
  const int*   coords = (const int*)d_in[1];
  // params flattened in alphabetical key order starting at d_in[3]:
  const float* b_in  = (const float*)d_in[3];
  const float* b_out = (const float*)d_in[4];
  const float* g_in  = (const float*)d_in[5];
  const float* g_out = (const float*)d_in[6];
  const float* w_in  = (const float*)d_in[80];
  const float* w_out = (const float*)d_in[81];

  const size_t V1 = (size_t)33 * 200 * 200;

  char* ws = (char*)d_ws;
  size_t off = 0;
  auto take = [&](size_t bytes) -> void* {
    void* p = ws + off; off = (off + bytes + 255) & ~(size_t)255; return p;
  };
  u16*   bufT  = (u16*)take(V1 * 16 * sizeof(u16));     // current activation (bf16)
  u16*   bufR  = (u16*)take(V1 * 16 * sizeof(u16));     // temp / stem dense (bf16)
  float* F     = (float*)take(V1 * 16 * sizeof(float)); // conv accumulator (f32)
  float* maskA = (float*)take(V1 * sizeof(float));
  float* maskB = (float*)take(V1 * sizeof(float));
  u16*   wpk   = (u16*)take((size_t)1 << 21);           // packed weights (max ~0.9MB)
  float* stats = (float*)take(257 * sizeof(float));

  auto zero_f = [&](float* p, size_t n) {
    k_zero_f32<<<(unsigned)((n + 255) / 256), 256, 0, stream>>>(p, n);
  };
  auto zero_h = [&](u16* p, size_t n) {
    k_zero_u16<<<(unsigned)((n + 255) / 256), 256, 0, stream>>>(p, n);
  };

  auto conv = [&](const u16* in, Dims di, const float* w, const float* bias, Dims dd,
                  int kd, int kh, int kw, int sd, int sh, int sw,
                  int pd, int ph, int pw, const float* omask, float* out) {
    int segTiles = (kw * di.C + 31) >> 5;
    int NT = dd.C >> 4;
    long long tot = (long long)kd * kh * segTiles * NT * 512;
    k_pack_w<<<(unsigned)((tot + 255) / 256), 256, 0, stream>>>(
        w, wpk, kd, kh, kw, di.C, dd.C, segTiles);
    ConvP P{di.D, di.H, di.W, di.C, dd.D, dd.H, dd.W, dd.C,
            kd, kh, kw, sd, sh, sw, pd, ph, pw, segTiles, NT};
    dim3 grid((unsigned)((dd.W + 15) >> 4), (unsigned)dd.H, (unsigned)dd.D);
    k_conv_wmma<<<grid, 32 * NT, 0, stream>>>(in, wpk, bias, omask, out, P);
  };

  auto bn = [&](const float* x, const float* mask, const float* g, const float* b,
                const u16* resid, u16* out_bf, float* out_f32,
                size_t nvox, int C, int relu) {
    zero_f(stats, (size_t)(2 * C + 1));
    size_t tot = nvox * (size_t)C;
    size_t nb = (tot + 255) / 256; if (nb > 4096) nb = 4096;
    k_bn_stats<<<(unsigned)nb, 256, 0, stream>>>(x, mask, stats, nvox, C);
    k_bn_apply<<<(unsigned)((tot + 255) / 256), 256, 0, stream>>>(
        x, mask, stats, g, b, resid, out_bf, out_f32, nvox, C, relu, 1e-3f);
  };

  auto maskdown = [&](const float* mi, float* mo, Dims di, Dims dd,
                      int kd, int kh, int kw, int sd, int sh, int sw,
                      int pd, int ph, int pw) {
    size_t n = (size_t)dd.D * dd.H * dd.W;
    k_mask_down<<<(unsigned)((n + 255) / 256), 256, 0, stream>>>(
        mi, mo, di.D, di.H, di.W, dd.D, dd.H, dd.W,
        kd, kh, kw, sd, sh, sw, pd, ph, pw);
  };

  auto basic_block = [&](Dims d, const float* mask, int pidx) {
    const float* bb1 = (const float*)d_in[pidx + 0];
    const float* bb2 = (const float*)d_in[pidx + 1];
    const float* c1b = (const float*)d_in[pidx + 2];
    const float* c2b = (const float*)d_in[pidx + 3];
    const float* gg1 = (const float*)d_in[pidx + 4];
    const float* gg2 = (const float*)d_in[pidx + 5];
    const float* w1  = (const float*)d_in[pidx + 6];
    const float* w2  = (const float*)d_in[pidx + 7];
    size_t nv = (size_t)d.D * d.H * d.W;
    conv(bufT, d, w1, c1b, d, 3, 3, 3, 1, 1, 1, 1, 1, 1, mask, F);
    bn(F, mask, gg1, bb1, nullptr, bufR, nullptr, nv, d.C, 1);
    conv(bufR, d, w2, c2b, d, 3, 3, 3, 1, 1, 1, 1, 1, 1, mask, F);
    bn(F, mask, gg2, bb2, /*resid=*/bufT, bufT, nullptr, nv, d.C, 1);
  };

  // ---- stem: scatter -> dense bf16 + mask, conv 5->16, BN+ReLU ----
  Dims d0{33, 200, 200, 5}, d1{33, 200, 200, 16};
  zero_h(bufR, V1 * 5);
  zero_f(maskA, V1);
  k_scatter<<<(60000 + 255) / 256, 256, 0, stream>>>(feats, coords, bufR, maskA,
                                                     60000, 200, 200, 5);
  conv(bufR, d0, w_in, nullptr, d1, 3, 3, 3, 1, 1, 1, 1, 1, 1, maskA, F);
  bn(F, maskA, g_in, b_in, nullptr, bufT, nullptr, V1, 16, 1);

  // ---- stage 1: two basic blocks @16ch, full res ----
  basic_block(d1, maskA, 7);
  basic_block(d1, maskA, 15);

  // ---- stages 2..4: downsample conv + BN+ReLU + two blocks ----
  struct Stage { Dims din, dout; int pd, ph, pw; int ibd, igd, iwd, blk0, blk1; };
  Stage st[3] = {
    { {33, 200, 200, 16}, {17, 100, 100, 32}, 1, 1, 1, 23, 24, 25, 26, 34 },
    { {17, 100, 100, 32}, { 9,  50,  50, 64}, 1, 1, 1, 42, 43, 44, 45, 53 },
    { { 9,  50,  50, 64}, { 4,  25,  25,128}, 0, 1, 1, 61, 62, 63, 64, 72 },
  };
  float* mcur = maskA;
  float* mnext = maskB;
  for (int s = 0; s < 3; ++s) {
    const float* bd = (const float*)d_in[st[s].ibd];
    const float* gd = (const float*)d_in[st[s].igd];
    const float* wd = (const float*)d_in[st[s].iwd];
    Dims di = st[s].din, dd = st[s].dout;
    size_t nv = (size_t)dd.D * dd.H * dd.W;
    maskdown(mcur, mnext, di, dd, 3, 3, 3, 2, 2, 2, st[s].pd, st[s].ph, st[s].pw);
    conv(bufT, di, wd, nullptr, dd, 3, 3, 3, 2, 2, 2,
         st[s].pd, st[s].ph, st[s].pw, mnext, F);
    bn(F, mnext, gd, bd, nullptr, bufT, nullptr, nv, dd.C, 1);
    basic_block(dd, mnext, st[s].blk0);
    basic_block(dd, mnext, st[s].blk1);
    float* tmp = mcur; mcur = mnext; mnext = tmp;
  }

  // ---- head: (3,1,1) conv stride (2,1,1) pad 0, mask, BN+ReLU -> f32 out ----
  Dims d4{4, 25, 25, 128}, dof{1, 25, 25, 128};
  maskdown(mcur, mnext, d4, dof, 3, 1, 1, 2, 1, 1, 0, 0, 0);
  conv(bufT, d4, w_out, nullptr, dof, 3, 1, 1, 2, 1, 1, 0, 0, 0, mnext, F);
  bn(F, mnext, g_out, b_out, nullptr, nullptr, (float*)d_out,
     (size_t)25 * 25, 128, 1);
}
